// MegaNeRF_49220325212767
// MI455X (gfx1250) — matrix operand order repaired and checked by reading, not compiled, not run on hardware
//
#include <hip/hip_runtime.h>
#include <hip/hip_bf16.h>

typedef __attribute__((ext_vector_type(16))) _Float16 v16h;
typedef __attribute__((ext_vector_type(8)))  float    v8f;

#define NPTS 16384
#define NEXP 8
#define TILEM 64                    // points per workgroup
#define MAXTILES (NPTS / TILEM + NEXP)   // 264 (expert padding)
#define NPERM (NPTS + NEXP * TILEM)      // 16896

// ---------------- workspace layout (bytes) ----------------
constexpr size_t SZ_W0   = (size_t)NEXP*256*128*2;   // layer0: 111->128 pad, transposed
constexpr size_t SZ_W256 = (size_t)NEXP*256*256*2;   // 256x256 layers
constexpr size_t SZ_W4   = (size_t)NEXP*256*384*2;   // skip layer: 367->384 pad
constexpr size_t SZ_WC1  = (size_t)NEXP*128*288*2;   // color1: 283->288 pad

constexpr size_t OFF_W0  = 0;
constexpr size_t OFF_W1  = OFF_W0 + SZ_W0;
constexpr size_t OFF_W2  = OFF_W1 + SZ_W256;
constexpr size_t OFF_W3  = OFF_W2 + SZ_W256;
constexpr size_t OFF_W4  = OFF_W3 + SZ_W256;
constexpr size_t OFF_W5  = OFF_W4 + SZ_W4;
constexpr size_t OFF_W6  = OFF_W5 + SZ_W256;
constexpr size_t OFF_W7  = OFF_W6 + SZ_W256;
constexpr size_t OFF_WF  = OFF_W7 + SZ_W256;
constexpr size_t OFF_WC1 = OFF_WF + SZ_W256;
constexpr size_t OFF_X0  = OFF_WC1 + SZ_WC1;            // f16 [N][128]
constexpr size_t OFF_DE  = OFF_X0 + (size_t)NPTS*128*2; // f16 [N][32]
constexpr size_t OFF_WB  = OFF_DE + (size_t)NPTS*32*2;  // f32 [N]
constexpr size_t OFF_AS  = OFF_WB + (size_t)NPTS*4;     // i32 [N]
constexpr size_t OFF_FG  = OFF_AS + (size_t)NPTS*4;     // i32 [N]
constexpr size_t OFF_PERM= OFF_FG + (size_t)NPTS*4;     // i32 [NPERM]
constexpr size_t OFF_CTRL= OFF_PERM + (size_t)NPERM*4;  // count[8],cursor[8],offs[8],pad,tileExpert[MAXTILES]

// ---------------- small utility kernels ----------------
__global__ void zero_ctrl(int* ctrl) {
    int t = threadIdx.x;
    if (t < 32) ctrl[t] = 0;    // count[8], cursor[8], offs[8], pad[8]
}

// W: f32 [E][Kin][OUT] -> Wt: f16 [E][OUT][Ktp], zero-padded rows Kin..Ktp
__global__ void repack_w(const float* __restrict__ W, _Float16* __restrict__ Wt,
                         int Kin, int OUT, int Ktp, int total) {
    for (int i = blockIdx.x * blockDim.x + threadIdx.x; i < total;
         i += gridDim.x * blockDim.x) {
        int k = i % Ktp;
        int r = i / Ktp;
        int n = r % OUT;
        int e = r / OUT;
        float v = (k < Kin) ? W[((size_t)e * Kin + k) * OUT + n] : 0.0f;
        Wt[i] = (_Float16)v;
    }
}

// ---------------- routing + encoding ----------------
__global__ void prep(const float* __restrict__ points, const float* __restrict__ viewdirs,
                     const int* __restrict__ appidx, const float* __restrict__ app_emb,
                     _Float16* __restrict__ X0, _Float16* __restrict__ DE,
                     float* __restrict__ WB, int* __restrict__ ASSIGN,
                     int* __restrict__ FG, int* __restrict__ count) {
    int i = blockIdx.x * blockDim.x + threadIdx.x;
    if (i >= NPTS) return;
    float px = points[3*i+0], py = points[3*i+1], pz = points[3*i+2];
    float vx = viewdirs[3*i+0], vy = viewdirs[3*i+1], vz = viewdirs[3*i+2];
    // centroids: x in {-75,-25,25,75}, y in {-50,50}, z=20
    const float cx[8] = {-75.f,-75.f,-25.f,-25.f, 25.f, 25.f, 75.f, 75.f};
    const float cy[8] = {-50.f, 50.f,-50.f, 50.f,-50.f, 50.f,-50.f, 50.f};
    int best = 0; float bd = 3.4e38f;
    #pragma unroll
    for (int e = 0; e < 8; e++) {
        float dx = px - cx[e], dy = py - cy[e], dz = pz - 20.f;
        float d2 = dx*dx + dy*dy + dz*dz;
        if (d2 < bd) { bd = d2; best = e; }
    }
    WB[i] = expf(-sqrtf(bd) * 0.1f);
    FG[i] = (px >= -80.f && px <= 80.f && py >= -80.f && py <= 80.f &&
             pz >= -4.f  && pz <= 44.f) ? 1 : 0;
    ASSIGN[i] = best;
    atomicAdd(&count[best], 1);

    _Float16* x0 = X0 + (size_t)i * 128;
    x0[0] = (_Float16)px; x0[1] = (_Float16)py; x0[2] = (_Float16)pz;
    int o = 3;
    #pragma unroll
    for (int j = 0; j < 10; j++) {
        float f = exp2f((float)j * (10.0f / 9.0f));
        x0[o++] = (_Float16)sinf(f*px); x0[o++] = (_Float16)sinf(f*py); x0[o++] = (_Float16)sinf(f*pz);
        x0[o++] = (_Float16)cosf(f*px); x0[o++] = (_Float16)cosf(f*py); x0[o++] = (_Float16)cosf(f*pz);
    }
    const float* emb = app_emb + ((size_t)best * 1000 + (size_t)appidx[i]) * 48;
    #pragma unroll 8
    for (int j = 0; j < 48; j++) x0[63 + j] = (_Float16)emb[j];
    for (int j = 111; j < 128; j++) x0[j] = (_Float16)0.0f;

    _Float16* de = DE + (size_t)i * 32;
    de[0] = (_Float16)vx; de[1] = (_Float16)vy; de[2] = (_Float16)vz;
    o = 3;
    #pragma unroll
    for (int j = 0; j < 4; j++) {
        float f = exp2f((float)j * (4.0f / 3.0f));
        de[o++] = (_Float16)sinf(f*vx); de[o++] = (_Float16)sinf(f*vy); de[o++] = (_Float16)sinf(f*vz);
        de[o++] = (_Float16)cosf(f*vx); de[o++] = (_Float16)cosf(f*vy); de[o++] = (_Float16)cosf(f*vz);
    }
    for (int j = 27; j < 32; j++) de[j] = (_Float16)0.0f;
}

__global__ void build_offsets(const int* __restrict__ count, int* __restrict__ offs,
                              int* __restrict__ tileExpert, int* __restrict__ perm) {
    int tid = threadIdx.x;
    for (int i = tid; i < NPERM; i += blockDim.x) perm[i] = -1;
    for (int t = tid; t < MAXTILES; t += blockDim.x) tileExpert[t] = -1;
    __syncthreads();
    if (tid == 0) {
        int run = 0;
        for (int e = 0; e < NEXP; e++) {
            offs[e] = run;
            int nt = (count[e] + TILEM - 1) / TILEM;
            int t0 = run / TILEM;
            for (int tt = 0; tt < nt; tt++) tileExpert[t0 + tt] = e;
            run += nt * TILEM;
        }
    }
}

__global__ void scatter(const int* __restrict__ assign, const int* __restrict__ offs,
                        int* __restrict__ cursor, int* __restrict__ perm) {
    int i = blockIdx.x * blockDim.x + threadIdx.x;
    if (i >= NPTS) return;
    int e = assign[i];
    int slot = atomicAdd(&cursor[e], 1);
    perm[offs[e] + slot] = i;
}

// ---------------- WMMA fragment helpers ----------------
union FragH { v16h h; uint4 u[2]; };

// A fragment: ISA 16-bit A 16x32 layout. row m, hi (=lane>>4):
//   halves 0..7  = K = kb + hi*8      .. +8
//   halves 8..15 = K = kb + 16 + hi*8 .. +8
// Split-K virtual row: k < split reads src0, else src1 (runs are 8-aligned, never straddle).
__device__ __forceinline__ v16h load_a_frag(const _Float16* src0, int ld0,
                                            const _Float16* src1, int ld1, int split,
                                            int m, int kb, int hi) {
    int k0 = kb + hi * 8;
    int k1 = kb + 16 + hi * 8;
    const _Float16* p0 = (k0 < split) ? (src0 + m * ld0 + k0) : (src1 + m * ld1 + (k0 - split));
    const _Float16* p1 = (k1 < split) ? (src0 + m * ld0 + k1) : (src1 + m * ld1 + (k1 - split));
    FragH f;
    f.u[0] = *(const uint4*)p0;
    f.u[1] = *(const uint4*)p1;
    return f.h;
}

// B fragment: 32x16 f16 B; lane n (=lane&15) holds column n, K = kb + hi*8 .. +8 and kb + 16 + hi*8 .. +8 (hi = lane>>4).
// Weights stored transposed [out][Ktp] so these are two 16B reads.
__device__ __forceinline__ v16h load_b_frag(const _Float16* Wt, int Kt, int col,
                                            int kb, int lane) {
    const _Float16* p = Wt + (size_t)col * Kt + kb + ((lane >> 4) * 8);
    FragH f;
    f.u[0] = *(const uint4*)p;
    f.u[1] = *(const uint4*)(p + 16);
    return f.h;
}

// gfx1250 global_prefetch warm-up of an upcoming weight block (L2/near caches)
__device__ __forceinline__ void prefetch_block(const void* p, int bytes, int tid) {
    const char* c = (const char*)p;
    for (int off = tid * 256; off < bytes; off += 256 * 256)
        __builtin_prefetch(c + off, 0, 1);
}

// One fused layer for a 64-point tile: out[64][OUTW] = act([in|in2] @ W + b)
// 8 waves; wave w owns column tiles w, w+8 (NT of them); 4 row tiles of 16.
template<int NT>
__device__ __forceinline__ void gemm_layer64(const _Float16* in0, int ld0,
                                             const _Float16* in1, int ld1, int split,
                                             const _Float16* Wt, int Kt,
                                             const float* bias,
                                             _Float16* out, int ldo,
                                             bool do_relu, int lane, int wave) {
    v8f acc[4][NT];
    #pragma unroll
    for (int rt = 0; rt < 4; rt++)
        #pragma unroll
        for (int t = 0; t < NT; t++)
            acc[rt][t] = (v8f){0.f,0.f,0.f,0.f,0.f,0.f,0.f,0.f};
    int m  = lane & 15;
    int hi = lane >> 4;
    for (int kb = 0; kb < Kt; kb += 32) {
        v16h b[NT];
        #pragma unroll
        for (int t = 0; t < NT; t++) {
            int cbase = (wave + t * 8) * 16;
            b[t] = load_b_frag(Wt, Kt, cbase + m, kb, lane);
        }
        #pragma unroll
        for (int rt = 0; rt < 4; rt++) {
            v16h a = load_a_frag(in0, ld0, in1, ld1, split, rt * 16 + m, kb, hi);
            #pragma unroll
            for (int t = 0; t < NT; t++)
                acc[rt][t] = __builtin_amdgcn_wmma_f32_16x16x32_f16(
                    false, a, false, b[t], (short)0, acc[rt][t], false, false);
        }
    }
    // C/D layout: lane&15 = N column, rows M = (lane>>4)*8 + r
    #pragma unroll
    for (int t = 0; t < NT; t++) {
        int n = (wave + t * 8) * 16 + m;
        float bv = bias[n];
        #pragma unroll
        for (int rt = 0; rt < 4; rt++) {
            #pragma unroll
            for (int r = 0; r < 8; r++) {
                float v = acc[rt][t][r] + bv;
                if (do_relu) v = v > 0.f ? v : 0.f;
                out[(rt * 16 + hi * 8 + r) * ldo + n] = (_Float16)v;
            }
        }
    }
}

// ---------------- main fused MoE-MLP kernel ----------------
struct MoEParams {
    const int* perm;
    const int* tileExpert;
    const _Float16* X0;
    const _Float16* DE;
    const float* WB;
    const int* FG;
    const _Float16* Wt[8];   // repacked f16 layers
    const _Float16* Wft;
    const _Float16* Wc1t;
    const float* bs[8];      // f32 biases (E,256) each
    const float* bf;
    const float* bc1;
    const float* Wd;         // (E,256,1) f32
    const float* bd;         // (E,1)
    const float* Wc2;        // (E,128,3) f32
    const float* bc2;        // (E,3)
    float* out;              // [N] density then [N,3] color
};

__global__ __launch_bounds__(256) void mega_moe(MoEParams P) {
    // ~120KB static LDS (CDNA5: up to 320KB per workgroup)
    __shared__ __align__(16) _Float16 bufA[TILEM * 384];
    __shared__ __align__(16) _Float16 bufB[TILEM * 384];
    __shared__ __align__(16) _Float16 x0s[TILEM * 128];
    __shared__ __align__(16) _Float16 des[TILEM * 32];
    __shared__ int   pidx[TILEM];
    __shared__ float wv[TILEM];
    __shared__ int   fgs[TILEM];
    __shared__ float sdens[TILEM];

    int tile = blockIdx.x;
    int e = P.tileExpert[tile];
    if (e < 0) return;
    int tid = threadIdx.x;
    int lane = tid & 31, wave = tid >> 5;

    if (tid < TILEM) {
        int pi = P.perm[tile * TILEM + tid];
        pidx[tid] = pi;
        wv[tid]   = (pi >= 0) ? P.WB[pi] : 0.f;
        fgs[tid]  = (pi >= 0) ? P.FG[pi] : 0;
    }
    __syncthreads();

    // cooperative load of x0 rows (64 x 128 f16): 32B chunks
    for (int i = tid; i < TILEM * 8; i += 256) {
        int row = i >> 3, seg = (i & 7) * 16;
        int pi = pidx[row];
        uint4* dst = (uint4*)(x0s + row * 128 + seg);
        if (pi >= 0) *dst = *(const uint4*)(P.X0 + (size_t)pi * 128 + seg);
        else { uint4 z; z.x = z.y = z.z = z.w = 0u; *dst = z; }
    }
    // dir-enc rows (64 x 32 f16): 32B chunks
    for (int i = tid; i < TILEM * 2; i += 256) {
        int row = i >> 1, seg = (i & 1) * 16;
        int pi = pidx[row];
        uint4* dst = (uint4*)(des + row * 32 + seg);
        if (pi >= 0) *dst = *(const uint4*)(P.DE + (size_t)pi * 32 + seg);
        else { uint4 z; z.x = z.y = z.z = z.w = 0u; *dst = z; }
    }
    __syncthreads();

    const int NOSPLIT = 1 << 30;
    const _Float16* W0 = P.Wt[0] + (size_t)e*256*128;
    const _Float16* W1 = P.Wt[1] + (size_t)e*256*256;
    const _Float16* W2 = P.Wt[2] + (size_t)e*256*256;
    const _Float16* W3 = P.Wt[3] + (size_t)e*256*256;
    const _Float16* W4 = P.Wt[4] + (size_t)e*256*384;
    const _Float16* W5 = P.Wt[5] + (size_t)e*256*256;
    const _Float16* W6 = P.Wt[6] + (size_t)e*256*256;
    const _Float16* W7 = P.Wt[7] + (size_t)e*256*256;
    const _Float16* WF = P.Wft   + (size_t)e*256*256;
    const _Float16* WC = P.Wc1t  + (size_t)e*128*288;

    // Layer 0: x0(128) -> bufB   (prefetch next layer while computing)
    prefetch_block(W1, 256*256*2, tid);
    gemm_layer64<2>(x0s, 128, nullptr, 0, NOSPLIT, W0, 128, P.bs[0] + e*256,
                    bufB, 384, true, lane, wave);
    __syncthreads();
    prefetch_block(W2, 256*256*2, tid);
    gemm_layer64<2>(bufB, 384, nullptr, 0, NOSPLIT, W1, 256, P.bs[1] + e*256,
                    bufA, 384, true, lane, wave);
    __syncthreads();
    prefetch_block(W3, 256*256*2, tid);
    gemm_layer64<2>(bufA, 384, nullptr, 0, NOSPLIT, W2, 256, P.bs[2] + e*256,
                    bufB, 384, true, lane, wave);
    __syncthreads();
    prefetch_block(W4, 256*384*2, tid);
    gemm_layer64<2>(bufB, 384, nullptr, 0, NOSPLIT, W3, 256, P.bs[3] + e*256,
                    bufA, 384, true, lane, wave);
    __syncthreads();
    // Layer 4 (skip): virtual input [bufA(256) | x0s(128)], Kt=384
    prefetch_block(W5, 256*256*2, tid);
    gemm_layer64<2>(bufA, 384, x0s, 128, 256, W4, 384, P.bs[4] + e*256,
                    bufB, 384, true, lane, wave);
    __syncthreads();
    prefetch_block(W6, 256*256*2, tid);
    gemm_layer64<2>(bufB, 384, nullptr, 0, NOSPLIT, W5, 256, P.bs[5] + e*256,
                    bufA, 384, true, lane, wave);
    __syncthreads();
    prefetch_block(W7, 256*256*2, tid);
    gemm_layer64<2>(bufA, 384, nullptr, 0, NOSPLIT, W6, 256, P.bs[6] + e*256,
                    bufB, 384, true, lane, wave);
    __syncthreads();
    prefetch_block(WF, 256*256*2, tid);
    gemm_layer64<2>(bufB, 384, nullptr, 0, NOSPLIT, W7, 256, P.bs[7] + e*256,
                    bufA, 384, true, lane, wave);
    __syncthreads();

    // density head (VALU): 256-dot per point, from final hidden x in bufA
    if (tid < TILEM) {
        const float* wd = P.Wd + (size_t)e * 256;
        float s = 0.f;
        for (int k = 0; k < 256; k++) s += (float)bufA[tid * 384 + k] * wd[k];
        s += P.bd[e];
        sdens[tid] = s > 0.f ? s : 0.f;
    }
    // feature layer (no relu): bufA -> bufB
    prefetch_block(WC, 128*288*2, tid);
    gemm_layer64<2>(bufA, 384, nullptr, 0, NOSPLIT, WF, 256, P.bf + e*256,
                    bufB, 384, false, lane, wave);
    __syncthreads();
    // append dir-enc into bufB cols 256..287
    for (int i = tid; i < TILEM * 2; i += 256) {
        int row = i >> 1, seg = (i & 1) * 16;
        *(uint4*)(bufB + row * 384 + 256 + seg) = *(const uint4*)(des + row * 32 + seg);
    }
    __syncthreads();
    // color hidden: [feat|de](288) -> h(128) in bufA cols 0..127, relu
    // 8 waves x 1 column tile each covers OUTW=128
    gemm_layer64<1>(bufB, 384, nullptr, 0, NOSPLIT, WC, 288, P.bc1 + e*128,
                    bufA, 384, true, lane, wave);
    __syncthreads();

    // final outputs
    if (tid < TILEM) {
        int pi = pidx[tid];
        if (pi >= 0)
            P.out[pi] = fgs[tid] ? sdens[tid] * wv[tid] : 0.01f;
    }
    if (tid < TILEM * 3) {
        int pt = tid / 3, ch = tid % 3;
        int pi = pidx[pt];
        if (pi >= 0) {
            const float* wc2 = P.Wc2 + (size_t)e * 128 * 3 + ch;
            float s = 0.f;
            for (int k = 0; k < 128; k++) s += (float)bufA[pt * 384 + k] * wc2[k * 3];
            s += P.bc2[e * 3 + ch];
            float c = 1.f / (1.f + expf(-s));
            float w = wv[pt];
            float outc = fgs[pt] ? (c * w) / w : 0.8f;
            P.out[NPTS + (size_t)pi * 3 + ch] = outc;
        }
    }
}

// ---------------- launcher ----------------
extern "C" void kernel_launch(void* const* d_in, const int* in_sizes, int n_in,
                              void* d_out, int out_size, void* d_ws, size_t ws_size,
                              hipStream_t stream) {
    const float* points   = (const float*)d_in[0];
    const float* viewdirs = (const float*)d_in[1];
    const int*   appidx   = (const int*)  d_in[2];
    const float* app_emb  = (const float*)d_in[3];
    const float* Ws[8]; for (int i = 0; i < 8; i++) Ws[i] = (const float*)d_in[4 + i];
    const float* bs[8]; for (int i = 0; i < 8; i++) bs[i] = (const float*)d_in[12 + i];
    const float* Wd  = (const float*)d_in[20];
    const float* bd  = (const float*)d_in[21];
    const float* Wf  = (const float*)d_in[22];
    const float* bf  = (const float*)d_in[23];
    const float* Wc1 = (const float*)d_in[24];
    const float* bc1 = (const float*)d_in[25];
    const float* Wc2 = (const float*)d_in[26];
    const float* bc2 = (const float*)d_in[27];

    char* ws = (char*)d_ws;
    _Float16* W0t  = (_Float16*)(ws + OFF_W0);
    _Float16* W1t  = (_Float16*)(ws + OFF_W1);
    _Float16* W2t  = (_Float16*)(ws + OFF_W2);
    _Float16* W3t  = (_Float16*)(ws + OFF_W3);
    _Float16* W4t  = (_Float16*)(ws + OFF_W4);
    _Float16* W5t  = (_Float16*)(ws + OFF_W5);
    _Float16* W6t  = (_Float16*)(ws + OFF_W6);
    _Float16* W7t  = (_Float16*)(ws + OFF_W7);
    _Float16* Wft  = (_Float16*)(ws + OFF_WF);
    _Float16* Wc1t = (_Float16*)(ws + OFF_WC1);
    _Float16* X0   = (_Float16*)(ws + OFF_X0);
    _Float16* DE   = (_Float16*)(ws + OFF_DE);
    float*    WB   = (float*)(ws + OFF_WB);
    int*      AS   = (int*)(ws + OFF_AS);
    int*      FG   = (int*)(ws + OFF_FG);
    int*      PERM = (int*)(ws + OFF_PERM);
    int*      CTRL = (int*)(ws + OFF_CTRL);
    int* count  = CTRL;
    int* cursor = CTRL + 8;
    int* offs   = CTRL + 16;
    int* tileE  = CTRL + 32;

    zero_ctrl<<<1, 32, 0, stream>>>(CTRL);

    // repack all weights -> transposed, K-padded f16
    auto launch_repack = [&](const float* W, _Float16* Wt, int Kin, int OUT, int Ktp) {
        int total = NEXP * OUT * Ktp;
        repack_w<<<(total + 255) / 256, 256, 0, stream>>>(W, Wt, Kin, OUT, Ktp, total);
    };
    launch_repack(Ws[0], W0t, 111, 256, 128);
    launch_repack(Ws[1], W1t, 256, 256, 256);
    launch_repack(Ws[2], W2t, 256, 256, 256);
    launch_repack(Ws[3], W3t, 256, 256, 256);
    launch_repack(Ws[4], W4t, 367, 256, 384);
    launch_repack(Ws[5], W5t, 256, 256, 256);
    launch_repack(Ws[6], W6t, 256, 256, 256);
    launch_repack(Ws[7], W7t, 256, 256, 256);
    launch_repack(Wf,    Wft, 256, 256, 256);
    launch_repack(Wc1,  Wc1t, 283, 128, 288);

    prep<<<(NPTS + 255) / 256, 256, 0, stream>>>(points, viewdirs, appidx, app_emb,
                                                 X0, DE, WB, AS, FG, count);
    build_offsets<<<1, 256, 0, stream>>>(count, offs, tileE, PERM);
    scatter<<<(NPTS + 255) / 256, 256, 0, stream>>>(AS, offs, cursor, PERM);

    MoEParams P;
    P.perm = PERM; P.tileExpert = tileE;
    P.X0 = X0; P.DE = DE; P.WB = WB; P.FG = FG;
    _Float16* Wts[8] = {W0t, W1t, W2t, W3t, W4t, W5t, W6t, W7t};
    for (int i = 0; i < 8; i++) { P.Wt[i] = Wts[i]; P.bs[i] = bs[i]; }
    P.Wft = Wft; P.Wc1t = Wc1t;
    P.bf = bf; P.bc1 = bc1;
    P.Wd = Wd; P.bd = bd; P.Wc2 = Wc2; P.bc2 = bc2;
    P.out = (float*)d_out;

    mega_moe<<<MAXTILES, 256, 0, stream>>>(P);
}
